// Linear4bit_36575941493419
// MI455X (gfx1250) — matrix operand order repaired and checked by reading, not compiled, not run on hardware
//
#include <hip/hip_runtime.h>

typedef __attribute__((ext_vector_type(16))) __bf16 v16bf;
typedef __attribute__((ext_vector_type(8)))  float  v8f;

#define T_TOK  8192
#define IN_F   4096
#define OUT_F  16384
#define BM     128
#define BN     128
#define BK     32
#define LDSS   40            // padded bf16 row stride (80 B, 16B-aligned chunks)
#define NKSTEP (IN_F / BK)   // 128

__device__ __constant__ float c_nf4[16] = {
    -1.0f, -0.6961928009986877f, -0.5250730514526367f, -0.39491748809814453f,
    -0.28444138169288635f, -0.18477343022823334f, -0.09105003625154495f, 0.0f,
    0.07958029955625534f, 0.16093020141124725f, 0.24611230194568634f,
    0.33791524171829224f, 0.44070982933044434f, 0.5626170039176941f,
    0.7229568362236023f, 1.0f};

// One v_perm_b32: D = {hi[31:16], lo[31:16]} = packed {bf16(lo), bf16(hi)} (RTZ)
__device__ __forceinline__ unsigned pack2(float lo, float hi) {
  return __builtin_amdgcn_perm(__float_as_uint(hi), __float_as_uint(lo),
                               0x07060302u);
}

__global__ __launch_bounds__(256) void nf4_wmma_gemm(
    const float* __restrict__ X, const int* __restrict__ CODES,
    const float* __restrict__ AMAX, const float* __restrict__ BIAS,
    float* __restrict__ OUT) {
  // double-buffered tiles: compute from buf while staging buf^1
  __shared__ __align__(16) unsigned short Alds[2][BM * LDSS];
  __shared__ __align__(16) unsigned short Blds[2][BN * LDSS];

  const int t    = threadIdx.x;
  const int lane = t & 31;
  const int lh   = lane & 15;    // position within 16-lane half
  const int sel  = lane >> 4;    // which half of the wave
  const int wave = t >> 5;
  const int m_off = (wave & 3) * 32;   // wave's M patch inside tile
  const int n_off = (wave >> 2) * 64;  // wave's N patch inside tile
  const int mbase = blockIdx.x * BM;
  const int nbase = blockIdx.y * BN;

  // NF4 codebook resident in lanes 0..15 (mirrored in 16..31);
  // lookup = ds_bpermute by code*4 (index bits [6:2] wrap at 32 -> no mask needed)
  const int tvi = __float_as_int(c_nf4[t & 15]);

  // global staging: thread t handles row (t>>1), K half (t&1)*16
  const int lrow = t >> 1;
  const int lk   = (t & 1) * 16;
  const float* xrow = X + (size_t)(mbase + lrow) * IN_F + lk;
  const int*   crow = CODES + (size_t)(nbase + lrow) * IN_F + lk;
  // (coff + k0) >> 6 == (coff >> 6) + (k0 >> 6): coff%64 in {0,16}, k0%64 in
  // {0,32} -> no carry.  Hoist the base pointer; k0>>6 folds into the imm offset.
  const float* arow = AMAX + (((size_t)(nbase + lrow) * IN_F + lk) >> 6);

  v8f acc[2][4];
#pragma unroll
  for (int mt = 0; mt < 2; ++mt)
#pragma unroll
    for (int nt = 0; nt < 4; ++nt) acc[mt][nt] = (v8f)0.0f;

  unsigned au[8], bu[8];

  auto stage_global = [&](int k0) {
    const float4* xp = (const float4*)(xrow + k0);
    float4 f0 = xp[0], f1 = xp[1], f2 = xp[2], f3 = xp[3];
    au[0] = pack2(f0.x, f0.y); au[1] = pack2(f0.z, f0.w);
    au[2] = pack2(f1.x, f1.y); au[3] = pack2(f1.z, f1.w);
    au[4] = pack2(f2.x, f2.y); au[5] = pack2(f2.z, f2.w);
    au[6] = pack2(f3.x, f3.y); au[7] = pack2(f3.z, f3.w);

    const int4* cp = (const int4*)(crow + k0);
    int4 c0 = cp[0], c1 = cp[1], c2 = cp[2], c3 = cp[3];
    const float s = arow[k0 >> 6];   // 16-chunk never crosses an absmax block
    auto dq = [&](int c) -> float {
      return __int_as_float(__builtin_amdgcn_ds_bpermute(c << 2, tvi)) * s;
    };
    bu[0] = pack2(dq(c0.x), dq(c0.y)); bu[1] = pack2(dq(c0.z), dq(c0.w));
    bu[2] = pack2(dq(c1.x), dq(c1.y)); bu[3] = pack2(dq(c1.z), dq(c1.w));
    bu[4] = pack2(dq(c2.x), dq(c2.y)); bu[5] = pack2(dq(c2.z), dq(c2.w));
    bu[6] = pack2(dq(c3.x), dq(c3.y)); bu[7] = pack2(dq(c3.z), dq(c3.w));
  };

  auto stage_lds = [&](int buf) {
    uint4* ap = (uint4*)&Alds[buf][lrow * LDSS + lk];
    ap[0] = make_uint4(au[0], au[1], au[2], au[3]);
    ap[1] = make_uint4(au[4], au[5], au[6], au[7]);
    uint4* bp = (uint4*)&Blds[buf][lrow * LDSS + lk];
    bp[0] = make_uint4(bu[0], bu[1], bu[2], bu[3]);
    bp[1] = make_uint4(bu[4], bu[5], bu[6], bu[7]);
  };

  auto compute = [&](int buf) {
    const unsigned short* Ab = Alds[buf];
    const unsigned short* Bb = Blds[buf];
    union FragU { v16bf v; uint4 q[2]; } afr[2], bfr[4];
    // A 16x32 bf16 fragment (ISA 7.12.2): lanes<16 hold K{0..7,16..23}, lanes>=16 +8
#pragma unroll
    for (int mt = 0; mt < 2; ++mt) {
      const int row = m_off + mt * 16 + lh;
      afr[mt].q[0] = *(const uint4*)&Ab[row * LDSS + sel * 8];
      afr[mt].q[1] = *(const uint4*)&Ab[row * LDSS + 16 + sel * 8];
    }
    // B 32x16 bf16 fragment: same K order as A: lanes<16 hold K{0..7,16..23}, lanes>=16 +8
#pragma unroll
    for (int nt = 0; nt < 4; ++nt) {
      const int col = n_off + nt * 16 + lh;
      bfr[nt].q[0] = *(const uint4*)&Bb[col * LDSS + sel * 8];
      bfr[nt].q[1] = *(const uint4*)&Bb[col * LDSS + 16 + sel * 8];
    }
#pragma unroll
    for (int mt = 0; mt < 2; ++mt)
#pragma unroll
      for (int nt = 0; nt < 4; ++nt)
        acc[mt][nt] = __builtin_amdgcn_wmma_f32_16x16x32_bf16(
            false, afr[mt].v, false, bfr[nt].v, (short)0, acc[mt][nt], false, false);
  };

  stage_global(0);
  stage_lds(0);
  __syncthreads();
  int buf = 0;
  for (int ks = 1; ks < NKSTEP; ++ks) {
    stage_global(ks * BK);    // prefetch next tile while WMMAs run
    compute(buf);             // consume current buffer
    stage_lds(buf ^ 1);       // fill the other buffer (no readers on it)
    __syncthreads();          // one barrier per K-step
    buf ^= 1;
  }
  compute(buf);

  // epilogue: C/D layout VGPR r <-> M = r (+8 for upper half-lanes); lanes <-> N
#pragma unroll
  for (int nt = 0; nt < 4; ++nt) {
    const int gn = nbase + n_off + nt * 16 + lh;
    const float b = BIAS[gn];
#pragma unroll
    for (int mt = 0; mt < 2; ++mt) {
      const int gm = mbase + m_off + mt * 16 + sel * 8;
      float* op = OUT + (size_t)gm * OUT_F + gn;
      v8f a = acc[mt][nt];
#pragma unroll
      for (int r = 0; r < 8; ++r) op[(size_t)r * OUT_F] = a[r] + b;
    }
  }
}

extern "C" void kernel_launch(void* const* d_in, const int* in_sizes, int n_in,
                              void* d_out, int out_size, void* d_ws, size_t ws_size,
                              hipStream_t stream) {
  (void)in_sizes; (void)n_in; (void)out_size; (void)d_ws; (void)ws_size;
  const float* x     = (const float*)d_in[0];
  const int*   codes = (const int*)d_in[1];
  const float* amax  = (const float*)d_in[2];
  const float* bias  = (const float*)d_in[3];
  float* out = (float*)d_out;
  // x fastest along M so consecutive blocks share the same codes stripe in L2
  dim3 grid(T_TOK / BM, OUT_F / BN);
  nf4_wmma_gemm<<<grid, dim3(256), 0, stream>>>(x, codes, amax, bias, out);
}
